// GINVirtual_node_59820304499026
// MI455X (gfx1250) — matrix operand order repaired and checked by reading, not compiled, not run on hardware
//
#include <hip/hip_runtime.h>

// ---------------- problem constants (from reference) ----------------
#define NN   100000
#define EE   600000
#define GGR  4096
#define DNF  37
#define HF   128
#define DEF  6
#define EHF  256
#define LLAY 3
#define MT   64            // rows (edges/nodes) per block = 4 WMMA M-subtiles

typedef __attribute__((ext_vector_type(16))) __bf16 v16bf;
typedef __attribute__((ext_vector_type(8)))  float  v8f;

union Frag {
    v16bf v;
    unsigned short u[16];
    uint4 q[2];
};

__device__ __forceinline__ unsigned short f2bf(float f) {
    unsigned int u = __float_as_uint(f);
    unsigned int r = u + 0x7FFFu + ((u >> 16) & 1u);   // round-to-nearest-even
    return (unsigned short)(r >> 16);
}

__device__ __forceinline__ v8f wmma_bf16(const Frag& a, const Frag& b, v8f c) {
    return __builtin_amdgcn_wmma_f32_16x16x32_bf16(false, a.v, false, b.v, (short)0, c, false, false);
}

// ---------------- weight repack: f32 [K,N] -> bf16 B-fragments ----------------
// frag layout: ((kt*(N/16) + nt)*32 + lane)*16 + j
//   lane<16 : col = nt*16+lane,    k = kt*32 + (j<8 ? j : 16+j-8)      (j = 0..15)
//   lane>=16: col = nt*16+lane-16, k = kt*32 + 8 + (j<8 ? j : 16+j-8)
__global__ void pack_w_kernel(const float* __restrict__ W, const float* __restrict__ g,
                              unsigned short* __restrict__ outp, int K, int N)
{
    int idx = blockIdx.x * blockDim.x + threadIdx.x;
    if (idx >= K * N) return;
    int j    = idx & 15;
    int lane = (idx >> 4) & 31;
    int frag = idx >> 9;
    int ntiles = N >> 4;
    int nt = frag % ntiles;
    int kt = frag / ntiles;
    int half = lane >> 4, l16 = lane & 15;
    int col = nt * 16 + l16;
    int k   = kt * 32 + half * 8 + (j < 8 ? j : 16 + (j - 8));
    float s = 1.0f;
    if (g) s = g[col] * rsqrtf(1.0f + 1e-5f);          // BN scale fold
    outp[idx] = f2bf(W[(size_t)k * N + col] * s);
}

__global__ void fold_bias_kernel(const float* __restrict__ b, const float* __restrict__ g,
                                 const float* __restrict__ bt, float* __restrict__ o, int n)
{
    int i = blockIdx.x * blockDim.x + threadIdx.x;
    if (i >= n) return;
    float s = g ? g[i] * rsqrtf(1.0f + 1e-5f) : 1.0f;
    o[i] = b[i] * s + (bt ? bt[i] : 0.0f);
}

// ---------------- small elementwise / scatter kernels ----------------
__global__ void atom_enc_kernel(const float* __restrict__ x, const float* __restrict__ aW,
                                const float* __restrict__ ab, float* __restrict__ h)
{
    int i = blockIdx.x * blockDim.x + threadIdx.x;
    if (i >= NN * HF) return;
    int n = i >> 7, c = i & 127;
    float acc = ab[c];
    const float* xr = x + (size_t)n * DNF;
    #pragma unroll
    for (int k = 0; k < DNF; ++k) acc += xr[k] * aW[k * HF + c];
    h[i] = fmaxf(acc, 0.0f);
}

__global__ void vn_init_kernel(const float* __restrict__ vnp, float* __restrict__ vn)
{
    int i = blockIdx.x * blockDim.x + threadIdx.x;
    if (i >= GGR * HF) return;
    vn[i] = vnp[i & 127];
}

__global__ void haug_kernel(const float* __restrict__ h, const float* __restrict__ vn,
                            const int* __restrict__ batch, float* __restrict__ h_aug,
                            float* __restrict__ agg)
{
    int i = blockIdx.x * blockDim.x + threadIdx.x;
    if (i >= NN * HF) return;
    int n = i >> 7, c = i & 127;
    h_aug[i] = h[i] + vn[(size_t)batch[n] * HF + c];
    agg[i] = 0.0f;
}

__global__ void copy_kernel(const float* __restrict__ a, float* __restrict__ b, int n)
{
    int i = blockIdx.x * blockDim.x + threadIdx.x;
    if (i < n) b[i] = a[i];
}

__global__ void zero_kernel(float* __restrict__ p, int n)
{
    int i = blockIdx.x * blockDim.x + threadIdx.x;
    if (i < n) p[i] = 0.0f;
}

__global__ void vn_scatter_kernel(const float* __restrict__ h_aug, const int* __restrict__ batch,
                                  float* __restrict__ vn_tmp)
{
    int i = blockIdx.x * blockDim.x + threadIdx.x;
    if (i >= NN * HF) return;
    int n = i >> 7, c = i & 127;
    atomicAdd(&vn_tmp[(size_t)batch[n] * HF + c], h_aug[i]);
}

__global__ void pool_scatter_kernel(const float* __restrict__ h, const int* __restrict__ batch,
                                    float* __restrict__ hsum, float* __restrict__ counts)
{
    int i = blockIdx.x * blockDim.x + threadIdx.x;
    if (i >= NN * HF) return;
    int n = i >> 7, c = i & 127;
    int g = batch[n];
    atomicAdd(&hsum[(size_t)g * HF + c], h[i]);
    if (c == 0) atomicAdd(&counts[g], 1.0f);
}

__global__ void pool_div_kernel(float* __restrict__ hsum, const float* __restrict__ counts)
{
    int i = blockIdx.x * blockDim.x + threadIdx.x;
    if (i >= GGR * HF) return;
    hsum[i] /= fmaxf(counts[i >> 7], 1.0f);
}

__global__ void lin128_kernel(const float* __restrict__ in, const float* __restrict__ W,
                              const float* __restrict__ b, float* __restrict__ out, int M)
{
    int i = blockIdx.x * blockDim.x + threadIdx.x;
    if (i >= M * HF) return;
    int g = i >> 7, c = i & 127;
    float acc = b[c];
    const float* row = in + (size_t)g * HF;
    #pragma unroll 8
    for (int k = 0; k < HF; ++k) acc += row[k] * W[k * HF + c];
    out[i] = acc;
}

__global__ void final_dot_kernel(const float* __restrict__ in, const float* __restrict__ W4,
                                 const float* __restrict__ b4, float* __restrict__ out, int M)
{
    int g = blockIdx.x * blockDim.x + threadIdx.x;
    if (g >= M) return;
    float acc = b4[0];
    const float* row = in + (size_t)g * HF;
    #pragma unroll 8
    for (int k = 0; k < HF; ++k) acc += row[k] * W4[k];
    out[g] = acc;
}

// ---------------- fused edge kernel ----------------
// Per block: 64 edges (4 WMMA M-subtiles). Stage1: e1 = relu(ea@eW1+b1) -> LDS bf16.
// Stage2: 8 waves (one 16-col N-tile each); per K-step load B frag ONCE, issue 4 WMMAs
// against the 4 A sub-tiles (4x B reuse). Stage3: fused GIN message + atomic scatter.
__global__ __launch_bounds__(256) void edge_fused_kernel(
    const float* __restrict__ edge_attr,
    const int* __restrict__ srcI, const int* __restrict__ dstI,
    const float* __restrict__ eW1, const float* __restrict__ eb1,
    const unsigned short* __restrict__ eW2p, const float* __restrict__ eb2,
    const float* __restrict__ h_aug, float* __restrict__ agg, int E)
{
    __shared__ __align__(16) float sW1[DEF * EHF];
    __shared__ float sb1[EHF];
    __shared__ float sEA[MT][DEF];
    __shared__ __align__(16) unsigned short sE1[MT][EHF + 8];

    const int t = threadIdx.x;
    const int ebase = blockIdx.x * MT;

    for (int i = t; i < DEF * EHF; i += 256) sW1[i] = eW1[i];
    sb1[t] = eb1[t];
    for (int i = t; i < MT * DEF; i += 256) {
        int r = i / DEF, c = i - r * DEF;
        int e = ebase + r;
        sEA[r][c] = (e < E) ? edge_attr[(size_t)e * DEF + c] : 0.0f;
    }
    __syncthreads();

    {   // e1 = relu(ea @ eW1 + b1): each thread 4 rows x 16 cols
        int rb = t >> 4;
        int cb = (t & 15) * 16;
        #pragma unroll
        for (int sub = 0; sub < 4; ++sub) {
            int row = sub * 16 + rb;
            float a0 = sEA[row][0], a1 = sEA[row][1], a2 = sEA[row][2];
            float a3 = sEA[row][3], a4 = sEA[row][4], a5 = sEA[row][5];
            #pragma unroll
            for (int j = 0; j < 16; ++j) {
                int c = cb + j;
                float acc = sb1[c]
                    + a0 * sW1[0 * EHF + c] + a1 * sW1[1 * EHF + c] + a2 * sW1[2 * EHF + c]
                    + a3 * sW1[3 * EHF + c] + a4 * sW1[4 * EHF + c] + a5 * sW1[5 * EHF + c];
                sE1[row][c] = f2bf(fmaxf(acc, 0.0f));
            }
        }
    }
    __syncthreads();

    const int wave = t >> 5, lane = t & 31;
    const int half = lane >> 4, l16 = lane & 15;
    v8f acc[4];
    #pragma unroll
    for (int s = 0; s < 4; ++s) acc[s] = (v8f){0.f,0.f,0.f,0.f,0.f,0.f,0.f,0.f};

    #pragma unroll
    for (int kt = 0; kt < EHF / 32; ++kt) {
        Frag b;
        const uint4* bp = (const uint4*)&eW2p[(size_t)((kt * 8 + wave) * 32 + lane) * 16];
        b.q[0] = bp[0];
        b.q[1] = bp[1];
        if (kt + 1 < EHF / 32)
            __builtin_prefetch(&eW2p[(size_t)(((kt + 1) * 8 + wave) * 32 + lane) * 16], 0, 1);
        int kb = kt * 32 + half * 8;
        #pragma unroll
        for (int sub = 0; sub < 4; ++sub) {
            Frag a;
            const unsigned short* arow = &sE1[sub * 16 + l16][0];
            a.q[0] = *(const uint4*)(arow + kb);
            a.q[1] = *(const uint4*)(arow + kb + 16);
            acc[sub] = wmma_bf16(a, b, acc[sub]);
        }
    }

    int col = wave * 16 + l16;
    float bias = eb2[col];
    #pragma unroll
    for (int sub = 0; sub < 4; ++sub) {
        #pragma unroll
        for (int r = 0; r < 8; ++r) {
            int m = sub * 16 + half * 8 + r;   // D layout: lanes 0-15 -> M=r, lanes 16-31 -> M=8+r
            int e = ebase + m;
            if (e < E) {
                int s = srcI[e], d = dstI[e];
                float msg = fmaxf(h_aug[(size_t)s * HF + col] + acc[sub][r] + bias, 0.0f);
                atomicAdd(&agg[(size_t)d * HF + col], msg);
            }
        }
    }
}

// ---------------- generic fused 2-layer MLP (WMMA, 64-row blocks) ----------------
// out = act2( act1( (alpha*X0 + X1) @ W1p + b1f ) @ W2p + b2f )
// K1 == 128, N1 in {128,256}, N2 == 128.
__global__ __launch_bounds__(256) void mlp2_kernel(
    const float* __restrict__ X0, const float* __restrict__ X1,
    const float* __restrict__ epsp,
    const unsigned short* __restrict__ W1p, const float* __restrict__ b1f,
    const unsigned short* __restrict__ W2p, const float* __restrict__ b2f,
    float* __restrict__ out, int M, int K1, int N1, int N2,
    int relu_mid, int relu_out)
{
    __shared__ __align__(16) unsigned short sA[MT][128 + 8];
    __shared__ __align__(16) unsigned short sT[MT][256 + 8];
    const int t = threadIdx.x;
    const int mbase = blockIdx.x * MT;
    const float alpha = epsp ? (1.0f + epsp[0]) : 1.0f;

    for (int i = t; i < MT * K1; i += 256) {        // stage A (bf16)
        int r = i / K1, c = i - r * K1;
        int g = mbase + r;
        float v = 0.0f;
        if (g < M) {
            v = alpha * X0[(size_t)g * K1 + c];
            if (X1) v += X1[(size_t)g * K1 + c];
        }
        sA[r][c] = f2bf(v);
    }
    __syncthreads();

    const int wave = t >> 5, lane = t & 31;
    const int half = lane >> 4, l16 = lane & 15;
    const int nt1 = N1 >> 4;

    for (int nt = wave; nt < nt1; nt += 8) {        // GEMM1: [64,K1]@[K1,N1]
        v8f acc[4];
        #pragma unroll
        for (int s = 0; s < 4; ++s) acc[s] = (v8f){0.f,0.f,0.f,0.f,0.f,0.f,0.f,0.f};
        for (int kt = 0; kt < K1 / 32; ++kt) {
            Frag b;
            const uint4* bp = (const uint4*)&W1p[(size_t)((kt * nt1 + nt) * 32 + lane) * 16];
            b.q[0] = bp[0];
            b.q[1] = bp[1];
            int kb = kt * 32 + half * 8;
            #pragma unroll
            for (int sub = 0; sub < 4; ++sub) {
                Frag a;
                const unsigned short* arow = &sA[sub * 16 + l16][0];
                a.q[0] = *(const uint4*)(arow + kb);
                a.q[1] = *(const uint4*)(arow + kb + 16);
                acc[sub] = wmma_bf16(a, b, acc[sub]);
            }
        }
        int col = nt * 16 + l16;
        float b1 = b1f[col];
        #pragma unroll
        for (int sub = 0; sub < 4; ++sub) {
            #pragma unroll
            for (int r = 0; r < 8; ++r) {
                float v = acc[sub][r] + b1;
                if (relu_mid) v = fmaxf(v, 0.0f);
                sT[sub * 16 + half * 8 + r][col] = f2bf(v);
            }
        }
    }
    __syncthreads();

    {                                               // GEMM2: [64,N1]@[N1,N2], N2==128
        int nt = wave;
        v8f acc[4];
        #pragma unroll
        for (int s = 0; s < 4; ++s) acc[s] = (v8f){0.f,0.f,0.f,0.f,0.f,0.f,0.f,0.f};
        for (int kt = 0; kt < N1 / 32; ++kt) {
            Frag b;
            const uint4* bp = (const uint4*)&W2p[(size_t)((kt * (N2 >> 4) + nt) * 32 + lane) * 16];
            b.q[0] = bp[0];
            b.q[1] = bp[1];
            int kb = kt * 32 + half * 8;
            #pragma unroll
            for (int sub = 0; sub < 4; ++sub) {
                Frag a;
                const unsigned short* arow = &sT[sub * 16 + l16][0];
                a.q[0] = *(const uint4*)(arow + kb);
                a.q[1] = *(const uint4*)(arow + kb + 16);
                acc[sub] = wmma_bf16(a, b, acc[sub]);
            }
        }
        int col = nt * 16 + l16;
        float b2 = b2f[col];
        #pragma unroll
        for (int sub = 0; sub < 4; ++sub) {
            #pragma unroll
            for (int r = 0; r < 8; ++r) {
                int g = mbase + sub * 16 + half * 8 + r;
                if (g < M) {
                    float v = acc[sub][r] + b2;
                    if (relu_out) v = fmaxf(v, 0.0f);
                    out[(size_t)g * N2 + col] = v;
                }
            }
        }
    }
}

// ---------------- host orchestration ----------------
extern "C" void kernel_launch(void* const* d_in, const int* in_sizes, int n_in,
                              void* d_out, int out_size, void* d_ws, size_t ws_size,
                              hipStream_t stream)
{
    (void)in_sizes; (void)n_in; (void)out_size; (void)ws_size;
    const float* x         = (const float*)d_in[0];
    const int*   edge_idx  = (const int*)  d_in[1];
    const float* edge_attr = (const float*)d_in[2];
    const int*   batch     = (const int*)  d_in[3];
    const float* aW  = (const float*)d_in[4];
    const float* ab  = (const float*)d_in[5];
    const float* vnp = (const float*)d_in[6];
    const float* eps = (const float*)d_in[7];
    const float* eW1 = (const float*)d_in[8];
    const float* eb1 = (const float*)d_in[9];
    const float* eW2 = (const float*)d_in[10];
    const float* eb2 = (const float*)d_in[11];
    const float* mW1 = (const float*)d_in[12];
    const float* mb1 = (const float*)d_in[13];
    const float* mg1 = (const float*)d_in[14];
    const float* mbt1= (const float*)d_in[15];
    const float* mW2 = (const float*)d_in[16];
    const float* mb2 = (const float*)d_in[17];
    const float* vW1 = (const float*)d_in[18];
    const float* vb1 = (const float*)d_in[19];
    const float* vg1 = (const float*)d_in[20];
    const float* vbt1= (const float*)d_in[21];
    const float* vW2 = (const float*)d_in[22];
    const float* vb2 = (const float*)d_in[23];
    const float* vg2 = (const float*)d_in[24];
    const float* vbt2= (const float*)d_in[25];
    const float* lW1 = (const float*)d_in[26];
    const float* lb1 = (const float*)d_in[27];
    const float* lW2 = (const float*)d_in[28];
    const float* lb2 = (const float*)d_in[29];
    const float* lW3 = (const float*)d_in[30];
    const float* lb3 = (const float*)d_in[31];
    const float* lW4 = (const float*)d_in[32];
    const float* lb4 = (const float*)d_in[33];

    const int* srcI = edge_idx;
    const int* dstI = edge_idx + EE;

    char* base = (char*)d_ws;
    size_t off = 0;
    auto alloc = [&](size_t bytes) -> void* {
        void* p = base + off;
        off = (off + bytes + 255) & ~(size_t)255;
        return p;
    };

    float* h      = (float*)alloc((size_t)NN * HF * 4);
    float* h_aug  = (float*)alloc((size_t)NN * HF * 4);
    float* agg    = (float*)alloc((size_t)NN * HF * 4);
    float* vn_cur = (float*)alloc((size_t)GGR * HF * 4);
    float* vn_tmp = (float*)alloc((size_t)GGR * HF * 4);
    float* hsum   = (float*)alloc((size_t)GGR * HF * 4);
    float* counts = (float*)alloc((size_t)GGR * 4);
    float* q1     = (float*)alloc((size_t)GGR * HF * 4);
    float* q2     = (float*)alloc((size_t)GGR * HF * 4);

    unsigned short* eW2p = (unsigned short*)alloc((size_t)LLAY * EHF * HF * 2);
    unsigned short* mW1p = (unsigned short*)alloc((size_t)LLAY * HF * EHF * 2);
    unsigned short* mW2p = (unsigned short*)alloc((size_t)LLAY * EHF * HF * 2);
    unsigned short* vW1p = (unsigned short*)alloc((size_t)2 * HF * EHF * 2);
    unsigned short* vW2p = (unsigned short*)alloc((size_t)2 * EHF * HF * 2);
    unsigned short* lW1p = (unsigned short*)alloc((size_t)HF * HF * 2);
    unsigned short* lW2p = (unsigned short*)alloc((size_t)HF * HF * 2);

    float* mb1f = (float*)alloc((size_t)LLAY * EHF * 4);
    float* vb1f = (float*)alloc((size_t)2 * EHF * 4);
    float* vb2f = (float*)alloc((size_t)2 * HF * 4);

    auto cdiv = [](int a, int b) { return (a + b - 1) / b; };

    // ---- repack weights (bf16 B-fragments), fold BN into W1/bias ----
    for (int l = 0; l < LLAY; ++l) {
        pack_w_kernel<<<cdiv(EHF * HF, 256), 256, 0, stream>>>(
            eW2 + (size_t)l * EHF * HF, nullptr, eW2p + (size_t)l * EHF * HF, EHF, HF);
        pack_w_kernel<<<cdiv(HF * EHF, 256), 256, 0, stream>>>(
            mW1 + (size_t)l * HF * EHF, mg1 + (size_t)l * EHF, mW1p + (size_t)l * HF * EHF, HF, EHF);
        pack_w_kernel<<<cdiv(EHF * HF, 256), 256, 0, stream>>>(
            mW2 + (size_t)l * EHF * HF, nullptr, mW2p + (size_t)l * EHF * HF, EHF, HF);
        fold_bias_kernel<<<1, EHF, 0, stream>>>(
            mb1 + (size_t)l * EHF, mg1 + (size_t)l * EHF, mbt1 + (size_t)l * EHF,
            mb1f + (size_t)l * EHF, EHF);
    }
    for (int l = 0; l < LLAY - 1; ++l) {
        pack_w_kernel<<<cdiv(HF * EHF, 256), 256, 0, stream>>>(
            vW1 + (size_t)l * HF * EHF, vg1 + (size_t)l * EHF, vW1p + (size_t)l * HF * EHF, HF, EHF);
        pack_w_kernel<<<cdiv(EHF * HF, 256), 256, 0, stream>>>(
            vW2 + (size_t)l * EHF * HF, vg2 + (size_t)l * HF, vW2p + (size_t)l * EHF * HF, EHF, HF);
        fold_bias_kernel<<<1, EHF, 0, stream>>>(
            vb1 + (size_t)l * EHF, vg1 + (size_t)l * EHF, vbt1 + (size_t)l * EHF,
            vb1f + (size_t)l * EHF, EHF);
        fold_bias_kernel<<<1, HF, 0, stream>>>(
            vb2 + (size_t)l * HF, vg2 + (size_t)l * HF, vbt2 + (size_t)l * HF,
            vb2f + (size_t)l * HF, HF);
    }
    pack_w_kernel<<<cdiv(HF * HF, 256), 256, 0, stream>>>(lW1, nullptr, lW1p, HF, HF);
    pack_w_kernel<<<cdiv(HF * HF, 256), 256, 0, stream>>>(lW2, nullptr, lW2p, HF, HF);

    // ---- atom encoder + VN init ----
    atom_enc_kernel<<<cdiv(NN * HF, 256), 256, 0, stream>>>(x, aW, ab, h);
    vn_init_kernel<<<cdiv(GGR * HF, 256), 256, 0, stream>>>(vnp, vn_cur);

    // ---- layers ----
    for (int l = 0; l < LLAY; ++l) {
        haug_kernel<<<cdiv(NN * HF, 256), 256, 0, stream>>>(h, vn_cur, batch, h_aug, agg);

        edge_fused_kernel<<<cdiv(EE, MT), 256, 0, stream>>>(
            edge_attr, srcI, dstI,
            eW1 + (size_t)l * DEF * EHF, eb1 + (size_t)l * EHF,
            eW2p + (size_t)l * EHF * HF, eb2 + (size_t)l * HF,
            h_aug, agg, EE);

        mlp2_kernel<<<cdiv(NN, MT), 256, 0, stream>>>(
            h_aug, agg, eps + l,
            mW1p + (size_t)l * HF * EHF, mb1f + (size_t)l * EHF,
            mW2p + (size_t)l * EHF * HF, mb2 + (size_t)l * HF,
            h, NN, HF, EHF, HF, /*relu_mid=*/1, /*relu_out=*/(l < LLAY - 1) ? 1 : 0);

        if (l < LLAY - 1) {
            copy_kernel<<<cdiv(GGR * HF, 256), 256, 0, stream>>>(vn_cur, vn_tmp, GGR * HF);
            vn_scatter_kernel<<<cdiv(NN * HF, 256), 256, 0, stream>>>(h_aug, batch, vn_tmp);
            mlp2_kernel<<<cdiv(GGR, MT), 256, 0, stream>>>(
                vn_tmp, nullptr, nullptr,
                vW1p + (size_t)l * HF * EHF, vb1f + (size_t)l * EHF,
                vW2p + (size_t)l * EHF * HF, vb2f + (size_t)l * HF,
                vn_cur, GGR, HF, EHF, HF, /*relu_mid=*/1, /*relu_out=*/1);
        }
    }

    // ---- global mean pool ----
    zero_kernel<<<cdiv(GGR * HF, 256), 256, 0, stream>>>(hsum, GGR * HF);
    zero_kernel<<<cdiv(GGR, 256), 256, 0, stream>>>(counts, GGR);
    pool_scatter_kernel<<<cdiv(NN * HF, 256), 256, 0, stream>>>(h, batch, hsum, counts);
    pool_div_kernel<<<cdiv(GGR * HF, 256), 256, 0, stream>>>(hsum, counts);

    // ---- readout: lW1,lW2 via WMMA, then lW3, then dot with lW4 ----
    mlp2_kernel<<<cdiv(GGR, MT), 256, 0, stream>>>(
        hsum, nullptr, nullptr, lW1p, lb1, lW2p, lb2,
        q1, GGR, HF, HF, HF, /*relu_mid=*/0, /*relu_out=*/0);
    lin128_kernel<<<cdiv(GGR * HF, 256), 256, 0, stream>>>(q1, lW3, lb3, q2, GGR);
    final_dot_kernel<<<cdiv(GGR, 256), 256, 0, stream>>>(q2, lW4, lb4, (float*)d_out, GGR);
}
